// SparseTransformerBlock_47425028883012
// MI455X (gfx1250) — matrix operand order repaired and checked by reading, not compiled, not run on hardware
//
#include <hip/hip_runtime.h>

// ---------------------------------------------------------------------------
// Types / helpers
// ---------------------------------------------------------------------------
typedef unsigned short u16;
typedef unsigned int   u32;

typedef __attribute__((ext_vector_type(16))) __bf16 v16bf;
typedef __attribute__((ext_vector_type(8)))  float  v8f;
typedef __attribute__((ext_vector_type(4)))  int    v4i;

struct alignas(16) U4   { u32 x, y, z, w; };
struct alignas(16) U8x32{ U4 a, b; };          // 32 bytes == v16bf
struct alignas(16) F4   { float x, y, z, w; };

#define DEV static __device__ __forceinline__

DEV u16 f2bf(float f){ u32 u = __builtin_bit_cast(u32, f); u += 0x7fffu + ((u >> 16) & 1u); return (u16)(u >> 16); }
DEV float bf2f(u16 h){ return __builtin_bit_cast(float, (u32)h << 16); }

// 16-bit A-matrix 16x32 lane layout (ISA 7.12.2):
// lane<16: M=lane, e[0..7]=K(kh*8+0..7), e[8..15]=K(16+kh*8+0..7), kh=lane>>4
DEV v16bf make_afrag(const u16* base /*row of >=32 bf16, 16B aligned*/, int khalf){
  const U4* p = (const U4*)base;
  U8x32 u; u.a = p[khalf]; u.b = p[2 + khalf];
  return __builtin_bit_cast(v16bf, u);
}
// B-matrix 32x16: lane<16 -> N=lane, e[0..7]=K(kh*8+0..7), e[8..15]=K(16+kh*8+0..7)  (Bt stored [N][K])
DEV v16bf make_bfrag(const u16* base /*row of >=32 bf16, 16B aligned*/, int khalf){
  const U4* p = (const U4*)base;
  U8x32 u; u.a = p[khalf]; u.b = p[2 + khalf];
  return __builtin_bit_cast(v16bf, u);
}
DEV v8f wmma_bf16(v16bf a, v16bf b, v8f c){
  return __builtin_amdgcn_wmma_f32_16x16x32_bf16(false, a, false, b, (short)0, c, false, false);
}

// ---- CDNA5 async global->LDS copy (ASYNCcnt) with safe fallback -----------
#if __has_builtin(__builtin_amdgcn_global_load_async_to_lds_b128) && \
    __has_builtin(__builtin_amdgcn_s_wait_asynccnt)
#define ASYNC_COPY 1
typedef __attribute__((address_space(1))) v4i gas1_v4i;
typedef __attribute__((address_space(3))) v4i gas3_v4i;
DEV void async_cp16(const u16* g, const u16* l){
  // flat->AS1: identity; flat LDS addr low 32 bits == LDS byte offset (ISA 10.2)
  __builtin_amdgcn_global_load_async_to_lds_b128(
      (gas1_v4i*)(size_t)g, (gas3_v4i*)(u32)(size_t)l, 0, 0);
}
DEV void async_wait(){ __builtin_amdgcn_s_wait_asynccnt(0); }
#else
#define ASYNC_COPY 0
#endif

// ---------------------------------------------------------------------------
// 1) mod = silu(c) @ w_ada + b_ada      (2 x 6144, K=1024)
// ---------------------------------------------------------------------------
__global__ __launch_bounds__(256) void k_mod(const float* __restrict__ c,
                                             const float* __restrict__ w_ada,
                                             const float* __restrict__ b_ada,
                                             float* __restrict__ mod){
  __shared__ float sc[1024];
  const int b = blockIdx.y, tid = threadIdx.x;
  for (int k = tid; k < 1024; k += 256){
    float v = c[b*1024 + k];
    sc[k] = v / (1.f + __expf(-v));
  }
  __syncthreads();
  const int j = blockIdx.x*256 + tid;
  float acc = b_ada[j];
  for (int k = 0; k < 1024; ++k) acc += sc[k] * w_ada[(size_t)k*6144 + j];
  mod[(size_t)b*6144 + j] = acc;
}

// ---------------------------------------------------------------------------
// 2) weight convert+transpose: f32 [K][N] -> bf16 [N][K]
// ---------------------------------------------------------------------------
__global__ __launch_bounds__(256) void k_cvt_t(const float* __restrict__ src,
                                               u16* __restrict__ dst, int K, int N){
  __shared__ float tile[32][33];
  const int tx = threadIdx.x & 31, ty = threadIdx.x >> 5;   // 32x8
  const int n0 = blockIdx.x*32, k0 = blockIdx.y*32;
#pragma unroll
  for (int i = 0; i < 4; ++i)
    tile[ty + i*8][tx] = src[(size_t)(k0 + ty + i*8)*N + n0 + tx];
  __syncthreads();
#pragma unroll
  for (int i = 0; i < 4; ++i){
    int n = n0 + ty + i*8;
    dst[(size_t)n*K + k0 + tx] = f2bf(tile[tx][ty + i*8]);
  }
}

// ---------------------------------------------------------------------------
// 3) LayerNorm(x) * (1+scale) + shift  -> bf16     (one block per token row)
// ---------------------------------------------------------------------------
__global__ __launch_bounds__(256) void k_ln_mod(const float* __restrict__ x,
                                                const float* __restrict__ mod,
                                                int sidx, int cidx,
                                                u16* __restrict__ out){
  __shared__ float red[16];
  __shared__ float stat[2];
  const int row = blockIdx.x;
  const int b = row >> 14;                 // T = 16384
  const int tid = threadIdx.x;
  const float* xr = x + (size_t)row*1024;
  F4 v = *(const F4*)(xr + tid*4);
  float s = v.x+v.y+v.z+v.w;
  float q = v.x*v.x + v.y*v.y + v.z*v.z + v.w*v.w;
#pragma unroll
  for (int mk = 16; mk >= 1; mk >>= 1){ s += __shfl_xor(s, mk, 32); q += __shfl_xor(q, mk, 32); }
  if ((tid & 31) == 0){ red[tid>>5] = s; red[8 + (tid>>5)] = q; }
  __syncthreads();
  if (tid == 0){
    float ts = 0.f, tq = 0.f;
#pragma unroll
    for (int i = 0; i < 8; ++i){ ts += red[i]; tq += red[8+i]; }
    float mu  = ts * (1.f/1024.f);
    float var = tq * (1.f/1024.f) - mu*mu;
    stat[0] = mu; stat[1] = rsqrtf(var + 1e-6f);
  }
  __syncthreads();
  const float mu = stat[0], rs = stat[1];
  const float* base = mod + (size_t)b*6144;
  F4 shv = *(const F4*)(base + sidx*1024 + tid*4);
  F4 scv = *(const F4*)(base + cidx*1024 + tid*4);
  float y0 = (v.x-mu)*rs*(1.f+scv.x)+shv.x;
  float y1 = (v.y-mu)*rs*(1.f+scv.y)+shv.y;
  float y2 = (v.z-mu)*rs*(1.f+scv.z)+shv.z;
  float y3 = (v.w-mu)*rs*(1.f+scv.w)+shv.w;
  u32* op = (u32*)(out + (size_t)row*1024 + tid*4);
  op[0] = (u32)f2bf(y0) | ((u32)f2bf(y1) << 16);
  op[1] = (u32)f2bf(y2) | ((u32)f2bf(y3) << 16);
}

// ---------------------------------------------------------------------------
// 4) bf16 WMMA GEMM:  acc = A[M][K] * Bt[N][K]^T + bias
//    Epilogues: plain f32 / bf16, optional GELU, or fused residual+gate:
//       Out[gm][gn] = xres[gm][gn] + gate[b][gn] * (acc + bias)   (N==1024)
//    block tile 128x256, 8 waves (2x4), wave tile 64x64 = 4x4 WMMA tiles.
//    K step 32, double-buffered LDS; async global->LDS when available.
// ---------------------------------------------------------------------------
template<bool OUT_BF16, bool GELU, bool ADDGATE>
__global__ __launch_bounds__(256) void k_gemm(const u16* __restrict__ A,
                                              const u16* __restrict__ Bt,
                                              const float* __restrict__ bias,
                                              void* __restrict__ Out,
                                              int N, int K,
                                              const float* __restrict__ xres,
                                              const float* __restrict__ modv,
                                              int gidx){
  __shared__ __align__(16) u16 As[2][128*40];
  __shared__ __align__(16) u16 Bs[2][256*40];
  const int tid = threadIdx.x;
  const int lane = tid & 31, w = tid >> 5;
  const int half = lane >> 4, nl = lane & 15;
  const int bm0 = blockIdx.y * 128, bn0 = blockIdx.x * 256;
  const int wm = (w >> 2) * 64, wn = (w & 3) * 64;

  v8f acc[4][4] = {};

  // cooperative staging: A tile 128x32 (2 x16B/thread), B tile 256x32 (4 x16B/thread)
  const int ar = tid >> 1, aseg = (tid & 1) * 16;
  const int brr = tid;                      // one 32-elem row per thread

#if ASYNC_COPY
  auto issue = [&](int k0, int buf){
    const u16* ap = A  + (size_t)(bm0 + ar)*K + k0 + aseg;
    const u16* bp = Bt + (size_t)(bn0 + brr)*K + k0;
    async_cp16(ap,      As[buf] + ar*40 + aseg);
    async_cp16(ap + 8,  As[buf] + ar*40 + aseg + 8);
    async_cp16(bp,      Bs[buf] + brr*40);
    async_cp16(bp + 8,  Bs[buf] + brr*40 + 8);
    async_cp16(bp + 16, Bs[buf] + brr*40 + 16);
    async_cp16(bp + 24, Bs[buf] + brr*40 + 24);
  };
#else
  U4 sa0, sa1, sb0, sb1, sb2, sb3;
  auto stage = [&](int k0){
    const u16* ap = A  + (size_t)(bm0 + ar)*K + k0 + aseg;
    sa0 = *(const U4*)(ap);
    sa1 = *(const U4*)(ap + 8);
    const u16* bp = Bt + (size_t)(bn0 + brr)*K + k0;
    sb0 = *(const U4*)(bp);
    sb1 = *(const U4*)(bp + 8);
    sb2 = *(const U4*)(bp + 16);
    sb3 = *(const U4*)(bp + 24);
  };
  auto commit = [&](int buf){
    *(U4*)(As[buf] + ar*40 + aseg)     = sa0;
    *(U4*)(As[buf] + ar*40 + aseg + 8) = sa1;
    u16* bq = Bs[buf] + brr*40;
    *(U4*)(bq)      = sb0;
    *(U4*)(bq + 8)  = sb1;
    *(U4*)(bq + 16) = sb2;
    *(U4*)(bq + 24) = sb3;
  };
#endif

  const int nT = K >> 5;
#if ASYNC_COPY
  issue(0, 0);
#else
  stage(0); commit(0);
#endif
  for (int t = 0; t < nT; ++t){
#if ASYNC_COPY
    async_wait();                 // own copies done; barrier covers the rest
    __syncthreads();
    if (t + 1 < nT) issue((t + 1) << 5, (t + 1) & 1);
#else
    __syncthreads();
    if (t + 1 < nT) stage((t + 1) << 5);
#endif
    const u16* as = As[t & 1];
    const u16* bs = Bs[t & 1];
    v16bf af[4];
#pragma unroll
    for (int mt = 0; mt < 4; ++mt)
      af[mt] = make_afrag(as + (wm + mt*16 + nl)*40, half);
#pragma unroll
    for (int nt = 0; nt < 4; ++nt){
      v16bf bf = make_bfrag(bs + (wn + nt*16 + nl)*40, half);
#pragma unroll
      for (int mt = 0; mt < 4; ++mt)
        acc[mt][nt] = wmma_bf16(af[mt], bf, acc[mt][nt]);
    }
#if !ASYNC_COPY
    if (t + 1 < nT) commit((t + 1) & 1);
#endif
  }

#pragma unroll
  for (int mt = 0; mt < 4; ++mt)
#pragma unroll
    for (int nt = 0; nt < 4; ++nt){
      const int gn = bn0 + wn + nt*16 + nl;
      const float bv = bias[gn];
#pragma unroll
      for (int r = 0; r < 8; ++r){
        const int gm = bm0 + wm + mt*16 + half*8 + r;
        float v = acc[mt][nt][r] + bv;
        if (GELU){
          float tg = 0.7978845608028654f * (v + 0.044715f*v*v*v);
          v = 0.5f*v*(1.0f + tanhf(tg));
        }
        if (ADDGATE){
          const int gb = gm >> 14;     // row -> batch (T = 16384, N == 1024)
          const float g = modv[(size_t)gb*6144 + gidx*1024 + gn];
          v = xres[(size_t)gm*N + gn] + g * v;
        }
        if (OUT_BF16) ((u16*)Out)[(size_t)gm*N + gn] = f2bf(v);
        else          ((float*)Out)[(size_t)gm*N + gn] = v;
      }
    }
}

// ---------------------------------------------------------------------------
// 5) Flash attention per (q-block 128, head, b*window). qkv bf16 [32768][3072]
//    cols = which*1024 + h*64 + d.  Out bf16 [32768][1024].
//    Softmax denominator accumulated via an extra "ones column" WMMA:
//    [O | l] = P @ [V | 1]  (same alpha-rescale recurrence as online softmax)
// ---------------------------------------------------------------------------
__global__ __launch_bounds__(256) void k_attn(const u16* __restrict__ qkv,
                                              u16* __restrict__ out){
  __shared__ __align__(16) u16 Kl[32*64];     // [key][feat]
  __shared__ __align__(16) u16 Vt[80*32];     // [feat][key], rows 64..79 = [1;0..0]
  __shared__ __align__(16) u16 Pl[8*16*32];   // per-wave [m][key]
  const int tid = threadIdx.x, lane = tid & 31, w = tid >> 5;
  const int half = lane >> 4, nl = lane & 15;
  const int qb = blockIdx.x;                  // 0..7   (128-row q blocks)
  const int h  = blockIdx.y;                  // 0..15
  const int bz = blockIdx.z;                  // b*16 + nw
  const int b = bz >> 4, nw = bz & 15;
  const size_t rowbase = (size_t)b*16384 + (size_t)nw*1024;

  // static ones/zeros rows of the extended V (written once)
  for (int idx = tid; idx < 16*32; idx += 256){
    const int rr = idx >> 5, kk = idx & 31;
    Vt[(64 + rr)*32 + kk] = (rr == 0) ? (u16)0x3F80 : (u16)0;   // bf16 1.0 / 0.0
  }

  // Q fragments (16 rows per wave), held in registers for the whole kv loop
  const int qrow = qb*128 + w*16 + nl;
  const u16* qrp = qkv + (rowbase + qrow)*3072 + h*64;
  const v16bf aq0 = make_afrag(qrp,      half);   // feats 0..31
  const v16bf aq1 = make_afrag(qrp + 32, half);   // feats 32..63

  float mrow[8];
  v8f o[5] = {};                               // o[4] accumulates l in column 0
#pragma unroll
  for (int r = 0; r < 8; ++r) mrow[r] = -1e30f;

  const int lkey = tid >> 3, lf0 = (tid & 7) * 8;
  u16* Pw = Pl + w*512;

  for (int j0 = 0; j0 < 1024; j0 += 32){
    __syncthreads();
    const u16* kvp = qkv + (rowbase + j0 + lkey)*3072 + h*64 + lf0;
    *(U4*)(Kl + lkey*64 + lf0) = *(const U4*)(kvp + 1024);      // K tile
    U4 vv = *(const U4*)(kvp + 2048);                            // V tile (transpose)
    const u16* vp = (const u16*)&vv;
#pragma unroll
    for (int i = 0; i < 8; ++i) Vt[(lf0 + i)*32 + lkey] = vp[i];
    __syncthreads();

    // S = Q K^T  (16 q-rows x 32 keys), scale 1/sqrt(64)
    v8f s0 = {}, s1 = {};
    {
      v16bf bk;
      bk = make_bfrag(Kl + nl*64,            half); s0 = wmma_bf16(aq0, bk, s0);
      bk = make_bfrag(Kl + (16 + nl)*64,     half); s1 = wmma_bf16(aq0, bk, s1);
      bk = make_bfrag(Kl + nl*64 + 32,       half); s0 = wmma_bf16(aq1, bk, s0);
      bk = make_bfrag(Kl + (16 + nl)*64 + 32,half); s1 = wmma_bf16(aq1, bk, s1);
    }

    // online softmax: only the row-max needs a cross-lane reduction
#pragma unroll
    for (int r = 0; r < 8; ++r){
      const float sc0 = s0[r] * 0.125f;
      const float sc1 = s1[r] * 0.125f;
      float rm = fmaxf(sc0, sc1);
#pragma unroll
      for (int mk = 1; mk < 16; mk <<= 1) rm = fmaxf(rm, __shfl_xor(rm, mk, 32));
      const float mn = fmaxf(mrow[r], rm);
      const float al = __expf(mrow[r] - mn);
      mrow[r] = mn;
#pragma unroll
      for (int f = 0; f < 5; ++f) o[f][r] *= al;
      const int m = half*8 + r;
      Pw[m*32 + nl]      = f2bf(__expf(sc0 - mn));
      Pw[m*32 + 16 + nl] = f2bf(__expf(sc1 - mn));
    }

    // [O | l] += P [V | 1]
    const v16bf ap = make_afrag(Pw + nl*32, half);
#pragma unroll
    for (int f = 0; f < 5; ++f){
      v16bf bv = make_bfrag(Vt + (f*16 + nl)*32, half);
      o[f] = wmma_bf16(ap, bv, o[f]);
    }
  }

  const int m0 = qb*128 + w*16;
#pragma unroll
  for (int r = 0; r < 8; ++r){
    const float lsum = __shfl(o[4][r], lane & 16, 32);   // column 0 of [P@1]
    const float rl = 1.0f / lsum;
    const int m = half*8 + r;
#pragma unroll
    for (int f = 0; f < 4; ++f)
      out[(rowbase + m0 + m)*1024 + h*64 + f*16 + nl] = f2bf(o[f][r] * rl);
  }
}

// ---------------------------------------------------------------------------
// Host launch
// ---------------------------------------------------------------------------
extern "C" void kernel_launch(void* const* d_in, const int* in_sizes, int n_in,
                              void* d_out, int out_size, void* d_ws, size_t ws_size,
                              hipStream_t stream){
  (void)in_sizes; (void)n_in; (void)out_size; (void)ws_size;
  const float* x      = (const float*)d_in[0];
  const float* c      = (const float*)d_in[1];
  const float* w_qkv  = (const float*)d_in[2];
  const float* b_qkv  = (const float*)d_in[3];
  const float* w_proj = (const float*)d_in[4];
  const float* b_proj = (const float*)d_in[5];
  const float* w_ada  = (const float*)d_in[6];
  const float* b_ada  = (const float*)d_in[7];
  const float* w1     = (const float*)d_in[8];
  const float* b1     = (const float*)d_in[9];
  const float* w2     = (const float*)d_in[10];
  const float* b2     = (const float*)d_in[11];
  float* out = (float*)d_out;

  const size_t ROWS = 32768;   // B*T
  char* wsp = (char*)d_ws;
  size_t off = 0;
  auto take = [&](size_t bytes)->char*{
    char* p = wsp + off; off = (off + bytes + 255) & ~(size_t)255; return p; };

  float* modb  = (float*)take((size_t)2*6144*4);
  u16* wqkvT   = (u16*)take((size_t)3072*1024*2);
  u16* wprojT  = (u16*)take((size_t)1024*1024*2);
  u16* w1T     = (u16*)take((size_t)4096*1024*2);
  u16* w2T     = (u16*)take((size_t)1024*4096*2);
  u16* hbuf    = (u16*)take(ROWS*1024*2);
  float* x1    = (float*)take(ROWS*1024*4);
  char* bigR   = take(ROWS*4096*2);                 // aliased region
  u16* qkvb    = (u16*)bigR;                        // 192 MB
  u16* attnout = (u16*)(bigR + ROWS*3072*2);        // 64 MB
  u16* midb    = (u16*)bigR;                        // 256 MB (after attn done)

  // weights -> bf16 transposed [N][K]
  k_cvt_t<<<dim3(3072/32, 1024/32), 256, 0, stream>>>(w_qkv,  wqkvT,  1024, 3072);
  k_cvt_t<<<dim3(1024/32, 1024/32), 256, 0, stream>>>(w_proj, wprojT, 1024, 1024);
  k_cvt_t<<<dim3(4096/32, 1024/32), 256, 0, stream>>>(w1,     w1T,    1024, 4096);
  k_cvt_t<<<dim3(1024/32, 4096/32), 256, 0, stream>>>(w2,     w2T,    4096, 1024);

  // adaLN modulation
  k_mod<<<dim3(24, 2), 256, 0, stream>>>(c, w_ada, b_ada, modb);

  // h = modulate(LN(x))  -> bf16
  k_ln_mod<<<(unsigned)ROWS, 256, 0, stream>>>(x, modb, 0, 1, hbuf);
  // qkv = h @ w_qkv + b_qkv  -> bf16
  k_gemm<true,  false, false><<<dim3(3072/256, ROWS/128), 256, 0, stream>>>(
      hbuf, wqkvT, b_qkv, qkvb, 3072, 1024, nullptr, nullptr, 0);
  // windowed flash attention -> bf16
  k_attn<<<dim3(8, 16, 32), 256, 0, stream>>>(qkvb, attnout);
  // x1 = x + gate_msa * (attnout @ w_proj + b_proj)   (fused epilogue)
  k_gemm<false, false, true ><<<dim3(1024/256, ROWS/128), 256, 0, stream>>>(
      attnout, wprojT, b_proj, x1, 1024, 1024, x, modb, 2);

  // h = modulate(LN(x1)) -> bf16
  k_ln_mod<<<(unsigned)ROWS, 256, 0, stream>>>(x1, modb, 3, 4, hbuf);
  // mid = gelu(h @ w1 + b1) -> bf16
  k_gemm<true,  true,  false><<<dim3(4096/256, ROWS/128), 256, 0, stream>>>(
      hbuf, w1T, b1, midb, 4096, 1024, nullptr, nullptr, 0);
  // out = x1 + gate_mlp * (mid @ w2 + b2)   (fused epilogue)
  k_gemm<false, false, true ><<<dim3(1024/256, ROWS/128), 256, 0, stream>>>(
      midb, w2T, b2, out, 1024, 4096, x1, modb, 5);
}
